// LayerNormGRUCell_34548716929790
// MI455X (gfx1250) — matrix operand order repaired and checked by reading, not compiled, not run on hardware
//
#include <hip/hip_runtime.h>
#include <hip/hip_bf16.h>

typedef __attribute__((ext_vector_type(16))) __bf16 v16bf;
typedef __attribute__((ext_vector_type(8)))  __bf16 v8bf;
typedef __attribute__((ext_vector_type(8)))  float  v8f;
typedef __attribute__((ext_vector_type(4)))  float  v4f;

#define HDIM 512
#define KDIM 512
#define MTILE 64          // rows per workgroup
#define PITCH 520         // LDS row pitch in bf16 elems (1040B: 16B aligned, bank-staggered)
#define LN_EPS 1e-5f

// ---------------------------------------------------------------------------
// Prepack Wi/Wh f32 [3,512,512] (g,h,i row-major) into bf16 B-fragment layout:
//   frag (g, ntile, kstep): lane L holds 16 contiguous K values
//   col = ntile*16 + (L&15), kstart = kstep*32 + (L>>4)*16
// ---------------------------------------------------------------------------
__global__ void gru_prepack_kernel(const float* __restrict__ Wi,
                                   const float* __restrict__ Wh,
                                   unsigned short* __restrict__ pkI_u,
                                   unsigned short* __restrict__ pkH_u) {
  __bf16* pkI = (__bf16*)pkI_u;
  __bf16* pkH = (__bf16*)pkH_u;
  int idx = blockIdx.x * blockDim.x + threadIdx.x;     // [0, 3*32*16*32)
  if (idx >= 3 * 32 * 16 * 32) return;
  int lane = idx & 31;
  int t    = idx >> 5;
  int ks   = t & 15;  t >>= 4;
  int ntg  = t & 31;
  int g    = t >> 5;
  int col  = ntg * 16 + (lane & 15);
  int krow = ks * 32 + (lane >> 4) * 16;
  size_t src = ((size_t)g * HDIM + col) * KDIM + krow;
  size_t dst = (size_t)idx * 16;
#pragma unroll
  for (int j = 0; j < 16; ++j) pkI[dst + j] = (__bf16)Wi[src + j];
#pragma unroll
  for (int j = 0; j < 16; ++j) pkH[dst + j] = (__bf16)Wh[src + j];
}

// ---------------------------------------------------------------------------
// helpers
// ---------------------------------------------------------------------------
__device__ __forceinline__ unsigned int pack_bf2(float a, float b) {
  unsigned short ua = __builtin_bit_cast(unsigned short, (__bf16)a);
  unsigned short ub = __builtin_bit_cast(unsigned short, (__bf16)b);
  return (unsigned int)ua | ((unsigned int)ub << 16);
}
__device__ __forceinline__ float unpack_bf2(unsigned int v, int hi) {
  unsigned short u = hi ? (unsigned short)(v >> 16) : (unsigned short)(v & 0xffffu);
  return (float)__builtin_bit_cast(__bf16, u);
}
__device__ __forceinline__ float sigmoidf_(float x) {
  return 1.0f / (1.0f + __expf(-x));
}

// ---------------------------------------------------------------------------
// One 64x512 GEMM pass over K=512 (bf16 WMMA) + bias + fused LayerNorm.
// Wave layout: 2(M) x 4(N); each wave: 2 Mtiles x 8 Ntiles of 16x16.
// ---------------------------------------------------------------------------
__device__ __forceinline__ void gemm_ln_pass(
    const __bf16* ldsA, const __bf16* __restrict__ Wpk,
    const float* __restrict__ bias, const float* __restrict__ lng,
    const float* __restrict__ lnb,
    int g, int lane, int wave_m, int wave_n,
    float* redS, float* redQ, v8f (&acc)[2][8]) {
  const int l16  = lane & 15;
  const int half = lane >> 4;

#pragma unroll
  for (int mt = 0; mt < 2; ++mt)
#pragma unroll
    for (int nt = 0; nt < 8; ++nt) { v8f z = {}; acc[mt][nt] = z; }

  // ---- K loop: 16 steps of 32, 16 WMMAs per step ----
  for (int ks = 0; ks < 16; ++ks) {
    v16bf afr[2];
#pragma unroll
    for (int mt = 0; mt < 2; ++mt) {
      int row = wave_m * 32 + mt * 16 + l16;
      const __bf16* p = ldsA + row * PITCH + ks * 32 + half * 16;
      v8bf lo = *(const v8bf*)(p);        // K {0..7}+16*half
      v8bf hi = *(const v8bf*)(p + 8);    // K {8..15}+16*half
#pragma unroll
      for (int j = 0; j < 8; ++j) { afr[mt][j] = lo[j]; afr[mt][8 + j] = hi[j]; }
    }
#pragma unroll
    for (int nt = 0; nt < 8; ++nt) {
      int ntg = wave_n * 8 + nt;
      const __bf16* bp = Wpk + ((((size_t)g * 32 + ntg) * 16 + ks) * 32 + lane) * 16;
      v16bf bfr = *(const v16bf*)bp;      // 32B/lane, coalesced
#pragma unroll
      for (int mt = 0; mt < 2; ++mt)
        acc[mt][nt] = __builtin_amdgcn_wmma_f32_16x16x32_bf16(
            false, afr[mt], false, bfr, (short)0, acc[mt][nt], false, false);
    }
  }

  // ---- bias + LN statistics ----
  float bcol[8], gcol[8], obcol[8];
#pragma unroll
  for (int nt = 0; nt < 8; ++nt) {
    int col = wave_n * 128 + nt * 16 + l16;
    bcol[nt]  = bias[g * HDIM + col];
    gcol[nt]  = lng[g * HDIM + col];
    obcol[nt] = lnb[g * HDIM + col];
  }
  float s[2][8], q[2][8];
#pragma unroll
  for (int mt = 0; mt < 2; ++mt)
#pragma unroll
    for (int v = 0; v < 8; ++v) { s[mt][v] = 0.f; q[mt][v] = 0.f; }
#pragma unroll
  for (int mt = 0; mt < 2; ++mt)
#pragma unroll
    for (int nt = 0; nt < 8; ++nt)
#pragma unroll
      for (int v = 0; v < 8; ++v) {
        float y = acc[mt][nt][v] + bcol[nt];
        acc[mt][nt][v] = y;
        s[mt][v] += y;
        q[mt][v] += y * y;
      }
  // reduce over the 16 column-lanes (lane bits 0..3); rows invariant
#pragma unroll
  for (int off = 1; off < 16; off <<= 1)
#pragma unroll
    for (int mt = 0; mt < 2; ++mt)
#pragma unroll
      for (int v = 0; v < 8; ++v) {
        s[mt][v] += __shfl_xor(s[mt][v], off, 32);
        q[mt][v] += __shfl_xor(q[mt][v], off, 32);
      }
  if (l16 == 0) {
#pragma unroll
    for (int mt = 0; mt < 2; ++mt)
#pragma unroll
      for (int v = 0; v < 8; ++v) {
        int row = wave_m * 32 + mt * 16 + half * 8 + v;
        redS[row * 4 + wave_n] = s[mt][v];
        redQ[row * 4 + wave_n] = q[mt][v];
      }
  }
  __syncthreads();
#pragma unroll
  for (int mt = 0; mt < 2; ++mt)
#pragma unroll
    for (int v = 0; v < 8; ++v) {
      int row = wave_m * 32 + mt * 16 + half * 8 + v;
      v4f ss = *(const v4f*)&redS[row * 4];
      v4f qq = *(const v4f*)&redQ[row * 4];
      float sum = (ss[0] + ss[1]) + (ss[2] + ss[3]);
      float sq  = (qq[0] + qq[1]) + (qq[2] + qq[3]);
      float mu  = sum * (1.0f / 512.0f);
      float var = sq * (1.0f / 512.0f) - mu * mu;
      float rstd = rsqrtf(var + LN_EPS);
#pragma unroll
      for (int nt = 0; nt < 8; ++nt)
        acc[mt][nt][v] = (acc[mt][nt][v] - mu) * rstd * gcol[nt] + obcol[nt];
    }
  __syncthreads();   // allow redS/redQ reuse by the next pass
}

// ---------------------------------------------------------------------------
// Fused LayerNorm-GRU cell: per block 64 rows, full gating fused, no global
// intermediates.
// ---------------------------------------------------------------------------
__global__ void __launch_bounds__(256, 1)
gru_fused_kernel(const float* __restrict__ x, const float* __restrict__ h,
                 const unsigned short* __restrict__ pkI_u,
                 const unsigned short* __restrict__ pkH_u,
                 const float* __restrict__ bi, const float* __restrict__ bh,
                 const float* __restrict__ ln_gi, const float* __restrict__ ln_bi,
                 const float* __restrict__ ln_gh, const float* __restrict__ ln_bh,
                 float* __restrict__ out) {
  extern __shared__ char smem[];
  __bf16* ldsX = (__bf16*)smem;                      // 64*520 bf16
  __bf16* ldsH = ldsX + MTILE * PITCH;               // 64*520 bf16
  float*  redS = (float*)(ldsH + MTILE * PITCH);     // 64*4 f32
  float*  redQ = redS + MTILE * 4;                   // 64*4 f32
  const __bf16* pkI = (const __bf16*)pkI_u;
  const __bf16* pkH = (const __bf16*)pkH_u;

  const int tid    = threadIdx.x;
  const int lane   = tid & 31;
  const int w      = tid >> 5;
  const int wave_m = w >> 2;   // 0..1
  const int wave_n = w & 3;    // 0..3
  const int l16    = lane & 15;
  const int half   = lane >> 4;
  const int brow   = blockIdx.x * MTILE;

  // ---- stage x/h row tiles into LDS as bf16 (convert on the fly) ----
  for (int u = tid; u < MTILE * KDIM / 4; u += 256) {
    int row = u >> 7;            // 128 float4 per row
    int c4  = (u & 127) * 4;
    v4f xv = *(const v4f*)&x[(size_t)(brow + row) * KDIM + c4];
    v4f hv = *(const v4f*)&h[(size_t)(brow + row) * KDIM + c4];
    __bf16* px = ldsX + row * PITCH + c4;
    __bf16* ph = ldsH + row * PITCH + c4;
#pragma unroll
    for (int j = 0; j < 4; ++j) { px[j] = (__bf16)xv[j]; ph[j] = (__bf16)hv[j]; }
  }
  __syncthreads();

  unsigned int r_pk[2][8][4];   // sigmoid(ai0+ah0), packed bf16
  unsigned int z_pk[2][8][4];   // sigmoid(ai1+ah1), packed bf16
  v8f aI[2][8], aH[2][8];

#pragma unroll 1
  for (int g = 0; g < 3; ++g) {
    gemm_ln_pass(ldsX, pkI, bi, ln_gi, ln_bi, g, lane, wave_m, wave_n, redS, redQ, aI);
    gemm_ln_pass(ldsH, pkH, bh, ln_gh, ln_bh, g, lane, wave_m, wave_n, redS, redQ, aH);

    if (g == 0) {
#pragma unroll
      for (int mt = 0; mt < 2; ++mt)
#pragma unroll
        for (int nt = 0; nt < 8; ++nt)
#pragma unroll
          for (int p = 0; p < 4; ++p)
            r_pk[mt][nt][p] = pack_bf2(sigmoidf_(aI[mt][nt][2 * p]     + aH[mt][nt][2 * p]),
                                       sigmoidf_(aI[mt][nt][2 * p + 1] + aH[mt][nt][2 * p + 1]));
    } else if (g == 1) {
#pragma unroll
      for (int mt = 0; mt < 2; ++mt)
#pragma unroll
        for (int nt = 0; nt < 8; ++nt)
#pragma unroll
          for (int p = 0; p < 4; ++p)
            z_pk[mt][nt][p] = pack_bf2(sigmoidf_(aI[mt][nt][2 * p]     + aH[mt][nt][2 * p]),
                                       sigmoidf_(aI[mt][nt][2 * p + 1] + aH[mt][nt][2 * p + 1]));
    } else {
      // n = tanh(ai2 + r*ah2);  out = (1-z)*n + z*h
#pragma unroll
      for (int mt = 0; mt < 2; ++mt)
#pragma unroll
        for (int nt = 0; nt < 8; ++nt)
#pragma unroll
          for (int v = 0; v < 8; ++v) {
            float r = unpack_bf2(r_pk[mt][nt][v >> 1], v & 1);
            float z = unpack_bf2(z_pk[mt][nt][v >> 1], v & 1);
            float n = tanhf(aI[mt][nt][v] + r * aH[mt][nt][v]);
            int row = brow + wave_m * 32 + mt * 16 + half * 8 + v;
            int col = wave_n * 128 + nt * 16 + l16;
            float hv = h[(size_t)row * HDIM + col];
            out[(size_t)row * HDIM + col] = (1.0f - z) * n + z * hv;
          }
    }
  }
}

// ---------------------------------------------------------------------------
// launcher
// ---------------------------------------------------------------------------
extern "C" void kernel_launch(void* const* d_in, const int* in_sizes, int n_in,
                              void* d_out, int out_size, void* d_ws, size_t ws_size,
                              hipStream_t stream) {
  const float* x     = (const float*)d_in[0];
  const float* h     = (const float*)d_in[1];
  const float* Wi    = (const float*)d_in[2];
  const float* bi    = (const float*)d_in[3];
  const float* Wh    = (const float*)d_in[4];
  const float* bh    = (const float*)d_in[5];
  const float* ln_gi = (const float*)d_in[6];
  const float* ln_bi = (const float*)d_in[7];
  const float* ln_gh = (const float*)d_in[8];
  const float* ln_bh = (const float*)d_in[9];
  float* out = (float*)d_out;

  unsigned short* pkI = (unsigned short*)d_ws;                 // 3*512*512 bf16
  unsigned short* pkH = pkI + (size_t)3 * HDIM * KDIM;         // 3*512*512 bf16

  const int B = in_sizes[0] / KDIM;                            // 32768

  // prepack weights into WMMA B-fragment layout (bf16)
  const int pk_threads = 3 * 32 * 16 * 32;                     // 49152
  gru_prepack_kernel<<<(pk_threads + 255) / 256, 256, 0, stream>>>(Wi, Wh, pkI, pkH);

  // fused GEMM + LN + gating
  size_t smem = (size_t)2 * MTILE * PITCH * sizeof(unsigned short)  // x/h tiles
              + (size_t)2 * MTILE * 4 * sizeof(float);              // LN reduction
  gru_fused_kernel<<<B / MTILE, 256, smem, stream>>>(
      x, h, pkI, pkH, bi, bh, ln_gi, ln_bi, ln_gh, ln_bh, out);
}
